// NeuromorphicNetwork_4818953306141
// MI455X (gfx1250) — compile-verified, matrix operand order checked by reading, not run on hardware
//
#include <hip/hip_runtime.h>
#include <hip/hip_bf16.h>

typedef __attribute__((ext_vector_type(16))) _Float16 v16h;
typedef __attribute__((ext_vector_type(8)))  _Float16 v8h;
typedef __attribute__((ext_vector_type(8)))  float    v8f;

#define BATCH      512
#define INPUT_DIM  4096
#define HIDDEN_DIM 8192
#define OUTPUT_DIM 1024
#define T_STEPS    10
#define LEAK       0.95f
#define REFRACT    1.0f
#define ADAPT      0.02f

// ---------------------------------------------------------------------------
// Phase 1: rate-encode inputs.  rate[b,i] = (#t : u_t < sigmoid(x)) / 10
// ---------------------------------------------------------------------------
__global__ void encode_spikes(const float* __restrict__ x,
                              float* __restrict__ rate, int n) {
    int i = blockIdx.x * blockDim.x + threadIdx.x;
    if (i >= n) return;
    float p = 1.0f / (1.0f + __expf(-x[i]));
    unsigned s = (unsigned)i * 2654435761u + 0x9e3779b9u;
    int cnt = 0;
#pragma unroll
    for (int t = 0; t < T_STEPS; ++t) {
        s ^= s << 13; s ^= s >> 17; s ^= s << 5;   // xorshift32
        float u = (float)(s >> 8) * (1.0f / 16777216.0f);
        cnt += (u < p) ? 1 : 0;
    }
    rate[i] = (float)cnt * 0.1f;
}

// ---------------------------------------------------------------------------
// WMMA GEMM: C[M,N] = A[M,K] * B[K,N], all f32 in memory, f16 WMMA compute.
// Block tile 128(M) x 64(N), BK = 32, 256 threads = 8 waves in 4x2 grid,
// each wave computes 32x32 via 2x2 v_wmma_f32_16x16x32_f16 accumulators.
// Dims must divide tiles (they do: 512/4096/8192/1024).
// ---------------------------------------------------------------------------
#define BM 128
#define BN 64
#define BK 32
#define PA 40   // LDS pitch (halves) for A rows: 80B, 16B aligned
#define PB 40   // LDS pitch (halves) for B columns

__global__ __launch_bounds__(256)
void gemm_wmma_f16(const float* __restrict__ A, const float* __restrict__ B,
                   float* __restrict__ C, int M, int N, int K) {
    __shared__ __align__(16) _Float16 sA[BM * PA];   // sA[row][k]
    __shared__ __align__(16) _Float16 sB[BN * PB];   // sB[col][k]  (transposed)

    const int tid  = threadIdx.x;
    const int lane = tid & 31;
    const int wave = tid >> 5;
    const int wm   = (wave & 3) * 32;   // wave M offset in block tile
    const int wn   = (wave >> 2) * 32;  // wave N offset in block tile

    const int m0 = blockIdx.y * BM;
    const int n0 = blockIdx.x * BN;

    // ISA 7.12.2 lane layout for 16-bit A/B fragments:
    // lanes 0-15:  row/col = lane,      K chunks {0-7, 16-23}
    // lanes 16-31: row/col = lane-16,   K chunks {8-15, 24-31}
    const int half_sel = lane >> 4;     // which half-wave
    const int lrow     = lane & 15;

    v8f acc[2][2] = {};

    for (int k0 = 0; k0 < K; k0 += BK) {
        // ---- prefetch next K tile into caches (global_prefetch_b8) ----
        if (k0 + BK < K) {
            __builtin_prefetch(A + (size_t)(m0 + (tid & 127)) * K + k0 + BK, 0, 1);
            __builtin_prefetch(B + (size_t)(k0 + BK + (tid & 31)) * N + n0, 0, 1);
        }

        // ---- stage A: 128x32 f32 -> f16, row-major ----
#pragma unroll
        for (int i = 0; i < 4; ++i) {
            int j   = tid + 256 * i;      // 0..1023 float4s
            int row = j >> 3;             // 8 float4 per 32-wide row
            int c4  = (j & 7) * 4;
            const float4 v = *(const float4*)(A + (size_t)(m0 + row) * K + k0 + c4);
            _Float16* d = &sA[row * PA + c4];
            d[0] = (_Float16)v.x; d[1] = (_Float16)v.y;
            d[2] = (_Float16)v.z; d[3] = (_Float16)v.w;
        }
        // ---- stage B: 32x64 f32 -> f16, transposed to sB[col][k] ----
#pragma unroll
        for (int i = 0; i < 2; ++i) {
            int j    = tid + 256 * i;     // 0..511 float4s
            int krow = j >> 4;            // 16 float4 per 64-wide row
            int n4   = (j & 15) * 4;
            const float4 v = *(const float4*)(B + (size_t)(k0 + krow) * N + n0 + n4);
            sB[(n4 + 0) * PB + krow] = (_Float16)v.x;
            sB[(n4 + 1) * PB + krow] = (_Float16)v.y;
            sB[(n4 + 2) * PB + krow] = (_Float16)v.z;
            sB[(n4 + 3) * PB + krow] = (_Float16)v.w;
        }
        __syncthreads();

        // ---- load fragments and issue WMMAs ----
        v16h afrag[2], bfrag[2];
#pragma unroll
        for (int sm = 0; sm < 2; ++sm) {
            const v8h* pa = (const v8h*)&sA[(wm + 16 * sm + lrow) * PA];
            v8h lo = pa[half_sel];        // K 0-7  or 8-15
            v8h hi = pa[half_sel + 2];    // K 16-23 or 24-31
            afrag[sm] = __builtin_shufflevector(lo, hi,
                0,1,2,3,4,5,6,7,8,9,10,11,12,13,14,15);
        }
#pragma unroll
        for (int sn = 0; sn < 2; ++sn) {
            const v8h* pb = (const v8h*)&sB[(wn + 16 * sn + lrow) * PB];
            v8h lo = pb[half_sel];
            v8h hi = pb[half_sel + 2];
            bfrag[sn] = __builtin_shufflevector(lo, hi,
                0,1,2,3,4,5,6,7,8,9,10,11,12,13,14,15);
        }
#pragma unroll
        for (int sm = 0; sm < 2; ++sm)
#pragma unroll
            for (int sn = 0; sn < 2; ++sn)
                acc[sm][sn] = __builtin_amdgcn_wmma_f32_16x16x32_f16(
                    false, afrag[sm], false, bfrag[sn],
                    (short)0, acc[sm][sn], false, false);
        __syncthreads();
    }

    // ---- epilogue: C/D layout — VGPR r: lanes 0-15 M=r, lanes 16-31 M=r+8 ----
#pragma unroll
    for (int sm = 0; sm < 2; ++sm) {
#pragma unroll
        for (int sn = 0; sn < 2; ++sn) {
            int row0 = m0 + wm + 16 * sm + half_sel * 8;
            int col  = n0 + wn + 16 * sn + lrow;
#pragma unroll
            for (int r = 0; r < 8; ++r)
                C[(size_t)(row0 + r) * N + col] = acc[sm][sn][r];
        }
    }
}

// ---------------------------------------------------------------------------
// LIF neuron recurrence: per-neuron state carried sequentially across the
// batch; parallel over neurons.  rate[b,n] = spikes(b,n)/10.
// ---------------------------------------------------------------------------
__global__ void lif_layer(const float* __restrict__ cur,
                          float* __restrict__ rate, int N, int B) {
    int n = blockIdx.x * blockDim.x + threadIdx.x;
    if (n >= N) return;
    float v = 0.0f, thr = 1.0f, last = -1e30f;
    int count = 0;
    for (int b = 0; b < B; ++b) {
        float I = cur[(size_t)b * N + n];
        int spikes = 0;
#pragma unroll
        for (int t = 0; t < T_STEPS; ++t) {
            v = v * LEAK + I;
            float tc = (float)count;
            bool sp = (v >= thr) && (tc - last > REFRACT);
            if (sp) {
                ++count;
                last = tc;
                v = 0.0f;
                if (count > 10) thr += ADAPT;
            }
            spikes += sp ? 1 : 0;
        }
        rate[(size_t)b * N + n] = (float)spikes * 0.1f;
    }
}

// ---------------------------------------------------------------------------
extern "C" void kernel_launch(void* const* d_in, const int* in_sizes, int n_in,
                              void* d_out, int out_size, void* d_ws, size_t ws_size,
                              hipStream_t stream) {
    const float* x    = (const float*)d_in[0];   // [512, 4096]
    const float* W_ih = (const float*)d_in[1];   // [4096, 8192]
    const float* W_ho = (const float*)d_in[2];   // [8192, 1024]
    float* out        = (float*)d_out;           // [512, 1024]

    // workspace layout (f32): 42 MB total
    float* R_in   = (float*)d_ws;                              // 512*4096
    float* cur_h  = R_in   + (size_t)BATCH * INPUT_DIM;        // 512*8192
    float* rate_h = cur_h  + (size_t)BATCH * HIDDEN_DIM;       // 512*8192
    float* cur_o  = rate_h + (size_t)BATCH * HIDDEN_DIM;       // 512*1024

    // 1) rate-encode inputs
    {
        int n = BATCH * INPUT_DIM;
        encode_spikes<<<(n + 255) / 256, 256, 0, stream>>>(x, R_in, n);
    }
    // 2) GEMM1: cur_h = R_in @ W_ih   (512x4096 * 4096x8192)
    {
        dim3 grid(HIDDEN_DIM / BN, BATCH / BM);
        gemm_wmma_f16<<<grid, 256, 0, stream>>>(R_in, W_ih, cur_h,
                                                BATCH, HIDDEN_DIM, INPUT_DIM);
    }
    // 3) hidden LIF dynamics -> rate_h
    lif_layer<<<HIDDEN_DIM / 256, 256, 0, stream>>>(cur_h, rate_h,
                                                    HIDDEN_DIM, BATCH);
    // 4) GEMM2: cur_o = rate_h @ W_ho (512x8192 * 8192x1024)
    {
        dim3 grid(OUTPUT_DIM / BN, BATCH / BM);
        gemm_wmma_f16<<<grid, 256, 0, stream>>>(rate_h, W_ho, cur_o,
                                                BATCH, OUTPUT_DIM, HIDDEN_DIM);
    }
    // 5) output LIF dynamics -> d_out (spike rate = mean over T)
    lif_layer<<<OUTPUT_DIM / 256, 256, 0, stream>>>(cur_o, out,
                                                    OUTPUT_DIM, BATCH);
}
